// PairEdgeLearnGNN_37993280701072
// MI455X (gfx1250) — matrix operand order repaired and checked by reading, not compiled, not run on hardware
//
#include <hip/hip_runtime.h>

#if defined(__has_builtin)
#  define HASB(x) __has_builtin(x)
#else
#  define HASB(x) 0
#endif

// ---------------- CDNA5 WMMA helpers (bf16 -> f32, 16x16x32) ----------------
typedef __attribute__((ext_vector_type(16))) __bf16 v16bf;
typedef __attribute__((ext_vector_type(8)))  float  v8f;

union Frag {
    v16bf v;
    unsigned short u[16];
    unsigned int   w[8];
    uint4 q[2];
};

// fp32 -> bf16 via native cast (backend may select v_cvt_pk_bf16_f32)
static __device__ __forceinline__ unsigned short f2bf(float f) {
    union { __bf16 h; unsigned short u; } cv;
    cv.h = (__bf16)f;
    return cv.u;
}

static __device__ __forceinline__ unsigned int f2bf2(float lo, float hi) {
#if HASB(__builtin_amdgcn_cvt_pk_bf16_f32)
    typedef __attribute__((ext_vector_type(2))) __bf16 v2bf;
    union { v2bf v; unsigned int u; } cv;
    cv.v = __builtin_amdgcn_cvt_pk_bf16_f32(lo, hi);
    return cv.u;
#else
    union { __bf16 h[2]; unsigned int u; } cv;
    cv.h[0] = (__bf16)lo;
    cv.h[1] = (__bf16)hi;
    return cv.u;
#endif
}

static __device__ __forceinline__ v8f wmma_bf16(const Frag& a, const Frag& b, v8f c) {
    return __builtin_amdgcn_wmma_f32_16x16x32_bf16(false, a.v, false, b.v,
                                                   (short)0, c, false, false);
}

static __device__ __forceinline__ v8f v8f_zero() {
    v8f z = {0.f,0.f,0.f,0.f,0.f,0.f,0.f,0.f};
    return z;
}

// async global->LDS staging of a 32KB tile (ASYNCcnt path), 16B/lane/issue
static __device__ __forceinline__ void async_stage_32k(const unsigned short* gsrc,
                                                       unsigned short* lds, int t) {
    unsigned long long gbase = (unsigned long long)gsrc;
    #pragma unroll
    for (int it = 0; it < 8; ++it) {
        int idx = it * 256 + t;                       // 2048 x 16B = 32KB
        unsigned long long ga = gbase + (unsigned long long)idx * 16ull;
        unsigned int la = (unsigned int)(unsigned long long)&lds[idx * 8];
        asm volatile("global_load_async_to_lds_b128 %0, %1, off"
                     :: "v"(la), "v"(ga) : "memory");
    }
    asm volatile("s_wait_asynccnt 0x0" ::: "memory");
}

// Problem constants
#define BB 64
#define NN 1024
#define DD 128

// ---------------- K0a: elementwise fp32 -> bf16 (count multiple of 2048) ----
__global__ __launch_bounds__(256) void k_cvt_bf16(const float* __restrict__ src,
                                                  unsigned short* __restrict__ dst) {
    long i = ((long)blockIdx.x * 256 + threadIdx.x) * 8;
    float4 f0 = *(const float4*)(src + i);
    float4 f1 = *(const float4*)(src + i + 4);
    uint4 o;
    o.x = f2bf2(f0.x, f0.y); o.y = f2bf2(f0.z, f0.w);
    o.z = f2bf2(f1.x, f1.y); o.w = f2bf2(f1.z, f1.w);
    *(uint4*)(dst + i) = o;
}

// ---------------- K0b: 128x128 fp32 -> transposed bf16 ----------------------
__global__ __launch_bounds__(256) void k_cvt_T(const float* __restrict__ src,
                                               unsigned short* __restrict__ dst) {
    int idx = blockIdx.x * 256 + threadIdx.x;   // 64 blocks -> 16384 elements
    int d = idx >> 7, e = idx & 127;
    dst[e * DD + d] = f2bf(src[idx]);
}

// ---------------- K1: s = xb @ wT (bf16 in, bf16 out), WMMA -----------------
// grid 512 (=B*N/128), block 256 (8 waves x 16 rows)
__global__ __launch_bounds__(256) void k_in_proj(const unsigned short* __restrict__ xb,
                                                 const unsigned short* __restrict__ wtb,
                                                 unsigned short* __restrict__ outb) {
    __shared__ unsigned short wT[DD * DD];      // wT[e][d] bf16 (32KB)
    int t = threadIdx.x;
    async_stage_32k(wtb, wT, t);
    __syncthreads();

    int wave = t >> 5, lane = t & 31, r = lane & 15, g = lane >> 4;
    long row0 = (long)blockIdx.x * 128 + wave * 16;

    v8f acc[8];
    #pragma unroll
    for (int i = 0; i < 8; ++i) acc[i] = v8f_zero();

    for (int ks = 0; ks < DD; ks += 32) {
        Frag a;
        const unsigned short* ap = xb + (row0 + r) * DD;
        a.q[0] = *(const uint4*)(ap + ks + g * 8);
        a.q[1] = *(const uint4*)(ap + ks + 16 + g * 8);
        #pragma unroll
        for (int tt = 0; tt < 8; ++tt) {
            Frag b;
            const uint4* bp = (const uint4*)(wT + (tt * 16 + r) * DD + ks + g * 8);
            b.q[0] = bp[0]; b.q[1] = bp[2];
            acc[tt] = wmma_bf16(a, b, acc[tt]);
        }
    }
    #pragma unroll
    for (int tt = 0; tt < 8; ++tt)
        #pragma unroll
        for (int v = 0; v < 8; ++v)
            outb[(row0 + v + 8 * g) * DD + tt * 16 + r] = f2bf(acc[tt][v]);
}

// ---------------- K2: scores = s1 @ s2^T (per batch), WMMA ------------------
// grid (8 jb, 8 ib, 64 b), block 256; s2 tile staged via async LDS DMA
__global__ __launch_bounds__(256) void k_scores(const unsigned short* __restrict__ s1b,
                                                const unsigned short* __restrict__ s2b,
                                                float* __restrict__ adjB) {
    __shared__ unsigned short s2t[128 * DD];    // s2 j-tile, row-major bf16 (32KB)
    int jb = blockIdx.x, ib = blockIdx.y, b = blockIdx.z;
    int t = threadIdx.x;

    async_stage_32k(s2b + ((long)b * NN + jb * 128) * DD, s2t, t);
    __syncthreads();

    int wave = t >> 5, lane = t & 31, r = lane & 15, g = lane >> 4;
    long row0 = (long)b * NN + ib * 128 + wave * 16;

    v8f acc[8];
    #pragma unroll
    for (int i = 0; i < 8; ++i) acc[i] = v8f_zero();

    for (int ks = 0; ks < DD; ks += 32) {
        Frag a;
        const unsigned short* ap = s1b + (row0 + r) * DD;
        a.q[0] = *(const uint4*)(ap + ks + g * 8);
        a.q[1] = *(const uint4*)(ap + ks + 16 + g * 8);
        #pragma unroll
        for (int tt = 0; tt < 8; ++tt) {
            Frag bfr;
            const uint4* bp = (const uint4*)(s2t + (tt * 16 + r) * DD + ks + g * 8);
            bfr.q[0] = bp[0]; bfr.q[1] = bp[2];
            acc[tt] = wmma_bf16(a, bfr, acc[tt]);
        }
    }
    #pragma unroll
    for (int tt = 0; tt < 8; ++tt)
        #pragma unroll
        for (int v = 0; v < 8; ++v)
            adjB[((long)b * NN + ib * 128 + wave * 16 + v + 8 * g) * NN
                 + jb * 128 + tt * 16 + r] = acc[tt][v];
}

// ---------------- K3: per-column online softmax stats (axis=1) --------------
// grid 256 (=B * N/256), block 256
__global__ __launch_bounds__(256) void k_colstats(const float* __restrict__ adjB,
                                                  float* __restrict__ mOut,
                                                  float* __restrict__ zOut) {
    int b = blockIdx.x >> 2;
    int j = ((blockIdx.x & 3) << 8) + threadIdx.x;
    const float* p = adjB + (long)b * NN * NN + j;
    float m = -3.4e38f, s = 0.f;
    for (int i = 0; i < NN; ++i) {
        __builtin_prefetch(p + (long)(i + 16) * NN, 0, 0);
        float v = p[(long)i * NN];
        float nm = fmaxf(m, v);
        s = s * __expf(m - nm) + __expf(v - nm);
        m = nm;
    }
    mOut[b * NN + j] = m;
    zOut[b * NN + j] = s;
}

// ------- K4: fused normalize->adj (in place) + h0=(adj+I)@x + gin linear ----
// grid 512 (b*8 + ib), block 256
__global__ __launch_bounds__(256) void k_h_gin(const float* __restrict__ x,
                                               const unsigned short* __restrict__ xb,
                                               const unsigned short* __restrict__ ginwb,
                                               const float* __restrict__ gin_b,
                                               const float* __restrict__ mcol,
                                               const float* __restrict__ zcol,
                                               float* __restrict__ adjB,
                                               float* __restrict__ h1) {
    __shared__ __align__(16) unsigned char smem[43520];
    unsigned short* xT   = (unsigned short*)smem;            // phase1: [128][40] bf16
    float*          mS   = (float*)(smem + 10240);           // phase1: 1024
    float*          zS   = (float*)(smem + 14336);           // phase1: 1024 (1/Z)
    unsigned short* h0s  = (unsigned short*)smem;            // phase2: [8][16][128] bf16
    unsigned short* ginS = (unsigned short*)(smem + 32768);  // phase2: [128][40] bf16

    int b = blockIdx.x >> 3, ib = blockIdx.x & 7;
    int t = threadIdx.x, wave = t >> 5, lane = t & 31, r = lane & 15, g = lane >> 4;
    int i0 = ib * 128 + wave * 16;

    for (int idx = t; idx < NN; idx += 256) {
        mS[idx] = mcol[b * NN + idx];
        zS[idx] = 1.0f / zcol[b * NN + idx];   // reciprocal once
    }

    v8f acc[8];
    #pragma unroll
    for (int i = 0; i < 8; ++i) acc[i] = v8f_zero();

    float* srow = adjB + ((long)b * NN + i0 + r) * NN;

    for (int j0 = 0; j0 < NN; j0 += 32) {
        __syncthreads();
        // stage xb[j0..j0+31][:] transposed to LDS xT[d][jj] (bf16 copies)
        #pragma unroll 4
        for (int q = 0; q < 8; ++q) {
            int idx = q * 256 + t;               // 2048 uint pairs
            int jj = idx >> 6;                   // 0..31
            int dp = (idx & 63) * 2;             // even d
            unsigned int val =
                *(const unsigned int*)(xb + ((long)b * NN + j0 + jj) * DD + dp);
            xT[dp * 40 + jj]       = (unsigned short)val;
            xT[(dp + 1) * 40 + jj] = (unsigned short)(val >> 16);
        }
        __syncthreads();

        // read score chunk, normalize -> adj (write back), bf16 A-frag
        int c1 = j0 + g * 8, c2 = j0 + 16 + g * 8;
        Frag a;
        #pragma unroll
        for (int kk = 0; kk < 8; kk += 2) {
            int j = c1 + kk;
            float a0 = __expf(srow[j]     - mS[j]    ) * zS[j];
            float a1 = __expf(srow[j + 1] - mS[j + 1]) * zS[j + 1];
            srow[j] = a0; srow[j + 1] = a1;
            a.w[kk >> 1] = f2bf2(a0, a1);
        }
        #pragma unroll
        for (int kk = 0; kk < 8; kk += 2) {
            int j = c2 + kk;
            float a0 = __expf(srow[j]     - mS[j]    ) * zS[j];
            float a1 = __expf(srow[j + 1] - mS[j + 1]) * zS[j + 1];
            srow[j] = a0; srow[j + 1] = a1;
            a.w[4 + (kk >> 1)] = f2bf2(a0, a1);
        }
        #pragma unroll
        for (int tt = 0; tt < 8; ++tt) {
            Frag bfr;
            const uint4* bp = (const uint4*)(xT + (tt * 16 + r) * 40 + g * 8);
            bfr.q[0] = bp[0]; bfr.q[1] = bp[2];
            acc[tt] = wmma_bf16(a, bfr, acc[tt]);
        }
    }
    __syncthreads();

    // identity (+x, fp32) and stage h0 tile (own wave region) as bf16
    #pragma unroll
    for (int tt = 0; tt < 8; ++tt)
        #pragma unroll
        for (int v = 0; v < 8; ++v) {
            float hv = acc[tt][v]
                     + x[((long)b * NN + i0 + v + 8 * g) * DD + tt * 16 + r];
            h0s[wave * 2048 + (v + 8 * g) * DD + tt * 16 + r] = f2bf(hv);
        }

    // h1 = h0 @ gin_w^T  (Bt[n=e][k=d] == gin_w row-major, pre-converted bf16)
    v8f acc2[8];
    #pragma unroll
    for (int i = 0; i < 8; ++i) acc2[i] = v8f_zero();

    for (int ks = 0; ks < DD; ks += 32) {
        __syncthreads();
        #pragma unroll 4
        for (int q = 0; q < 8; ++q) {
            int idx = q * 256 + t;               // 2048 uints
            int e = idx >> 4;                    // 0..127
            int kp = (idx & 15) * 2;             // even k within step
            *(unsigned int*)(ginS + e * 40 + kp) =
                *(const unsigned int*)(ginwb + e * DD + ks + kp);
        }
        __syncthreads();
        Frag a2;
        const unsigned short* hp = h0s + wave * 2048 + r * DD;
        a2.q[0] = *(const uint4*)(hp + ks + g * 8);
        a2.q[1] = *(const uint4*)(hp + ks + 16 + g * 8);
        #pragma unroll
        for (int tt = 0; tt < 8; ++tt) {
            Frag bfr;
            const uint4* bp = (const uint4*)(ginS + (tt * 16 + r) * 40 + g * 8);
            bfr.q[0] = bp[0]; bfr.q[1] = bp[2];
            acc2[tt] = wmma_bf16(a2, bfr, acc2[tt]);
        }
    }
    #pragma unroll
    for (int tt = 0; tt < 8; ++tt)
        #pragma unroll
        for (int v = 0; v < 8; ++v) {
            int e = tt * 16 + r;
            h1[((long)b * NN + i0 + v + 8 * g) * DD + e] = acc2[tt][v] + gin_b[e];
        }
}

// ---------------- K5: BN1 per-channel stats -> mu, rsqrt(var+eps) -----------
__global__ __launch_bounds__(256) void k_bn1stats(const float* __restrict__ h1,
                                                  float* __restrict__ mu,
                                                  float* __restrict__ rr) {
    __shared__ float r0[256], r1[256];
    int n = blockIdx.x, t = threadIdx.x;
    float s = 0.f, q = 0.f;
    for (int idx = t; idx < BB * DD; idx += 256) {
        int bb = idx >> 7, d = idx & 127;
        float v = h1[((long)bb * NN + n) * DD + d];
        s += v; q += v * v;
    }
    r0[t] = s; r1[t] = q; __syncthreads();
    for (int o = 128; o > 0; o >>= 1) {
        if (t < o) { r0[t] += r0[t + o]; r1[t] += r1[t + o]; }
        __syncthreads();
    }
    if (t == 0) {
        float m = r0[0] * (1.f / 8192.f);
        float var = r1[0] * (1.f / 8192.f) - m * m;
        mu[n] = m;
        rr[n] = rsqrtf(var + 1e-5f);
    }
}

// ---------------- K5b: fold BN1 affine into lc weights ----------------------
__global__ __launch_bounds__(256) void k_weff(const float* __restrict__ lc_w,
                                              const float* __restrict__ lc_b,
                                              const float* __restrict__ bn1_g,
                                              const float* __restrict__ bn1_b,
                                              const float* __restrict__ mu,
                                              const float* __restrict__ rr,
                                              float* __restrict__ weff,
                                              float* __restrict__ beff) {
    __shared__ float red0[256], red1[256];
    int t = threadIdx.x;
    float bp0 = 0.f, bp1 = 0.f;
    for (int n = t; n < NN; n += 256) {
        float gg = bn1_g[n] * rr[n];
        float cc = bn1_b[n] - mu[n] * gg;
        float w0 = lc_w[n], w1 = lc_w[NN + n];
        weff[n] = w0 * gg;
        weff[NN + n] = w1 * gg;
        bp0 += w0 * cc; bp1 += w1 * cc;
    }
    red0[t] = bp0; red1[t] = bp1; __syncthreads();
    for (int o = 128; o > 0; o >>= 1) {
        if (t < o) { red0[t] += red0[t + o]; red1[t] += red1[t + o]; }
        __syncthreads();
    }
    if (t == 0) { beff[0] = lc_b[0] + red0[0]; beff[1] = lc_b[1] + red1[0]; }
}

// ---------------- K6: lc contraction over N ---------------------------------
__global__ __launch_bounds__(128) void k_lc(const float* __restrict__ h1,
                                            const float* __restrict__ weff,
                                            const float* __restrict__ beff,
                                            float* __restrict__ lcout) {
    __shared__ float wl[2 * NN];
    int b = blockIdx.x, d = threadIdx.x;
    for (int idx = d; idx < 2 * NN; idx += 128) wl[idx] = weff[idx];
    __syncthreads();
    float a0 = beff[0], a1 = beff[1];
    for (int n = 0; n < NN; ++n) {
        float v = h1[((long)b * NN + n) * DD + d];
        a0 += wl[n] * v;
        a1 += wl[NN + n] * v;
    }
    lcout[(b * 2 + 0) * DD + d] = a0;
    lcout[(b * 2 + 1) * DD + d] = a1;
}

// ------- K7: conv13(same) + BN2 + ReLU + out linear + softmax -> probs ------
__global__ __launch_bounds__(256) void k_head(const float* __restrict__ lcout,
                                              const float* __restrict__ conv_w,
                                              const float* __restrict__ conv_b,
                                              const float* __restrict__ bn2_g,
                                              const float* __restrict__ bn2_b,
                                              const float* __restrict__ out_w,
                                              const float* __restrict__ out_b,
                                              float* __restrict__ probs) {
    __shared__ float y[BB * DD];
    __shared__ float red0[256], red1[256];
    __shared__ float cw[26];
    __shared__ float lg[128];
    __shared__ float stats[2];
    int t = threadIdx.x;
    if (t < 26) cw[t] = conv_w[t];
    __syncthreads();

    float s = 0.f, q = 0.f;
    for (int idx = t; idx < BB * DD; idx += 256) {
        int b = idx >> 7, d = idx & 127;
        float acc = conv_b[0];
        #pragma unroll
        for (int c = 0; c < 2; ++c)
            #pragma unroll
            for (int k = 0; k < 13; ++k) {
                int dd = d + k - 6;
                if (dd >= 0 && dd < DD)
                    acc += lcout[(b * 2 + c) * DD + dd] * cw[c * 13 + k];
            }
        y[idx] = acc; s += acc; q += acc * acc;
    }
    red0[t] = s; red1[t] = q; __syncthreads();
    for (int o = 128; o > 0; o >>= 1) {
        if (t < o) { red0[t] += red0[t + o]; red1[t] += red1[t + o]; }
        __syncthreads();
    }
    if (t == 0) {
        float m = red0[0] * (1.f / 8192.f);
        float var = red1[0] * (1.f / 8192.f) - m * m;
        stats[0] = m; stats[1] = rsqrtf(var + 1e-5f);
    }
    __syncthreads();
    float m = stats[0], rv = stats[1], g2 = bn2_g[0], b2 = bn2_b[0];
    for (int idx = t; idx < BB * DD; idx += 256)
        y[idx] = fmaxf((y[idx] - m) * rv * g2 + b2, 0.f);
    __syncthreads();
    if (t < 128) {
        int b = t >> 1, c = t & 1;
        float acc = out_b[c];
        for (int d = 0; d < DD; ++d) acc += y[b * DD + d] * out_w[c * DD + d];
        lg[t] = acc;
    }
    __syncthreads();
    if (t < 128) {
        int b = t >> 1, c = t & 1;
        float l0 = lg[b * 2], l1 = lg[b * 2 + 1];
        float mx = fmaxf(l0, l1);
        float e0 = __expf(l0 - mx), e1 = __expf(l1 - mx);
        probs[t] = ((c == 0) ? e0 : e1) / (e0 + e1);
    }
}

// ---------------- launch ----------------------------------------------------
extern "C" void kernel_launch(void* const* d_in, const int* in_sizes, int n_in,
                              void* d_out, int out_size, void* d_ws, size_t ws_size,
                              hipStream_t stream) {
    (void)in_sizes; (void)n_in; (void)out_size; (void)ws_size;
    const float* x      = (const float*)d_in[0];
    const float* w1     = (const float*)d_in[1];
    const float* w2     = (const float*)d_in[2];
    const float* gin_w  = (const float*)d_in[3];
    const float* gin_b  = (const float*)d_in[4];
    const float* bn1_g  = (const float*)d_in[5];
    const float* bn1_b  = (const float*)d_in[6];
    const float* lc_w   = (const float*)d_in[7];
    const float* lc_b   = (const float*)d_in[8];
    const float* conv_w = (const float*)d_in[9];
    const float* conv_b = (const float*)d_in[10];
    const float* bn2_g  = (const float*)d_in[11];
    const float* bn2_b  = (const float*)d_in[12];
    const float* out_w  = (const float*)d_in[13];
    const float* out_b  = (const float*)d_in[14];

    float* out  = (float*)d_out;
    float* adjB = out + BB * 2;   // probs (B*NC=128) first, then adj (B,N,N)

    // workspace layout (bytes)
    unsigned char* ws = (unsigned char*)d_ws;
    const size_t SZ_S = (size_t)BB * NN * DD * 2;      // 16 MB bf16
    unsigned short* s1b  = (unsigned short*)(ws);
    unsigned short* s2b  = (unsigned short*)(ws + SZ_S);
    unsigned short* xb   = (unsigned short*)(ws + 2 * SZ_S);
    float* mcol  = (float*)(ws + 3 * SZ_S);                        // B*N
    float* zcol  = (float*)(ws + 3 * SZ_S + (size_t)BB * NN * 4);
    float* h1    = (float*)(ws + 3 * SZ_S + (size_t)BB * NN * 8);  // B*N*D fp32
    unsigned char* p = ws + 3 * SZ_S + (size_t)BB * NN * 8 + (size_t)BB * NN * DD * 4;
    float* mu    = (float*)(p);                              // N
    float* rr    = (float*)(p + 4096);                       // N
    float* weff  = (float*)(p + 8192);                       // 2*N
    float* beff  = (float*)(p + 16384);                      // 2 (+pad)
    float* lcout = (float*)(p + 16640);                      // B*CF*D
    unsigned short* ginwb = (unsigned short*)(p + 82176);    // D*D bf16
    unsigned short* w1tb  = (unsigned short*)(p + 114944);   // D*D bf16 (transposed)
    unsigned short* w2tb  = (unsigned short*)(p + 147712);   // D*D bf16 (transposed)

    // one-shot conversions
    k_cvt_bf16<<<4096, 256, 0, stream>>>(x, xb);             // B*N*D
    k_cvt_bf16<<<8, 256, 0, stream>>>(gin_w, ginwb);         // D*D
    k_cvt_T<<<64, 256, 0, stream>>>(w1, w1tb);
    k_cvt_T<<<64, 256, 0, stream>>>(w2, w2tb);

    k_in_proj<<<512, 256, 0, stream>>>(xb, w1tb, s1b);
    k_in_proj<<<512, 256, 0, stream>>>(xb, w2tb, s2b);
    k_scores<<<dim3(8, 8, 64), 256, 0, stream>>>(s1b, s2b, adjB);
    k_colstats<<<256, 256, 0, stream>>>(adjB, mcol, zcol);
    k_h_gin<<<512, 256, 0, stream>>>(x, xb, ginwb, gin_b, mcol, zcol, adjB, h1);
    k_bn1stats<<<1024, 256, 0, stream>>>(h1, mu, rr);
    k_weff<<<1, 256, 0, stream>>>(lc_w, lc_b, bn1_g, bn1_b, mu, rr, weff, beff);
    k_lc<<<64, 128, 0, stream>>>(h1, weff, beff, lcout);
    k_head<<<1, 256, 0, stream>>>(lcout, conv_w, conv_b, bn2_g, bn2_b, out_w, out_b, out);
}
